// KeyPilotDecoder_28063316312423
// MI455X (gfx1250) — matrix operand order repaired and checked by reading, not compiled, not run on hardware
//
#include <hip/hip_runtime.h>
#include <hip/hip_bf16.h>
#include <math.h>

typedef _Float16 f16;
typedef _Float16 v16h __attribute__((ext_vector_type(16)));
typedef float    v8f  __attribute__((ext_vector_type(8)));

// ---------------------------------------------------------------------------
// WMMA fragment loaders (wave32 layouts per CDNA5 ISA 7.12.2)
//  A (16x32 f16, MxK): lane L -> m = L&15, g = L>>4.
//    halves 0..7  = A[m][kt*32 + 8g + 0..7]
//    halves 8..15 = A[m][kt*32 + 16 + 8g + 0..7]
//  B (32x16 f16, KxN): lane L -> n = L&15, g = L>>4; halves 0..7 = B[8g+0..7][n], 8..15 = B[16+8g+0..7][n]
//  C/D (16x16 f32): vgpr r -> m = r + 8*(L>>4), n = L&15
// ---------------------------------------------------------------------------
__device__ __forceinline__ v16h ld_frag_a(const f16* row, int g) {
  union { v16h v; float4 f[2]; } u;
  u.f[0] = *(const float4*)(row + 8 * g);
  u.f[1] = *(const float4*)(row + 16 + 8 * g);
  return u.v;
}
__device__ __forceinline__ v16h ld_frag_b(const f16* p, int ld, int g) {
  v16h v;
#pragma unroll
  for (int s = 0; s < 8; ++s) {
    v[s] = p[(8 * g + s) * ld];
    v[8 + s] = p[(16 + 8 * g + s) * ld];
  }
  return v;
}
__device__ __forceinline__ float gelu_exact(float x) {
  return 0.5f * x * (1.0f + erff(x * 0.70710678118654752f));
}
// LDS byte offset of a __shared__ object (flat LDS addr low 32 bits == LDS offset)
__device__ __forceinline__ unsigned lds_off(const void* p) {
  return (unsigned)(unsigned long long)p;
}
// Async global->LDS 16-byte copy (CDNA5, ASYNCcnt-tracked)
__device__ __forceinline__ void async_cp16(unsigned lds_addr, const void* gaddr) {
  asm volatile("global_load_async_to_lds_b128 %0, %1, off"
               :: "v"(lds_addr), "v"((unsigned long long)gaddr) : "memory");
}
__device__ __forceinline__ void async_wait0() {
  asm volatile("s_wait_asynccnt 0" ::: "memory");
}

// ---------------------------------------------------------------------------
// Generic tiled GEMM (f32 in/out): C[M,N] = act(A[M,K] @ W[K,N] + bias)
// 128x128 tile, 256 threads (8 waves), wave w owns m-tile w (8 n-tiles each).
// ---------------------------------------------------------------------------
__global__ __launch_bounds__(256) void gemm128(
    const float* __restrict__ A, const float* __restrict__ W,
    const float* __restrict__ bias, float* __restrict__ C,
    int M, int N, int K, int act)
{
  __shared__ __align__(16) f16 sA[128 * 32];   // [m][k], ld=32
  __shared__ __align__(16) f16 sB[32 * 128];   // [k][n], ld=128
  const int tid = threadIdx.x, wave = tid >> 5, lane = tid & 31;
  const int m0 = blockIdx.y * 128, n0 = blockIdx.x * 128;
  v8f acc[8] = {};
  const int nk = K >> 5;
  for (int kt = 0; kt < nk; ++kt) {
    __syncthreads();
    for (int e = tid; e < 128 * 32; e += 256) {
      int r = e >> 5, c = e & 31, m = m0 + r;
      sA[e] = (m < M) ? (f16)A[(size_t)m * K + kt * 32 + c] : (f16)0.f;
    }
    for (int e = tid; e < 32 * 128; e += 256) {
      int r = e >> 7, c = e & 127, n = n0 + c;
      sB[e] = (n < N) ? (f16)W[(size_t)(kt * 32 + r) * N + n] : (f16)0.f;
    }
    __syncthreads();
    v16h a = ld_frag_a(sA + (wave * 16 + (lane & 15)) * 32, lane >> 4);
#pragma unroll
    for (int nt = 0; nt < 8; ++nt) {
      v16h b = ld_frag_b(sB + (lane & 15) + nt * 16, 128, lane >> 4);
      acc[nt] = __builtin_amdgcn_wmma_f32_16x16x32_f16(
          false, a, false, b, (short)0, acc[nt], false, false);
    }
  }
  const int ml = 8 * (lane >> 4), nl = lane & 15;
#pragma unroll
  for (int nt = 0; nt < 8; ++nt) {
    int n = n0 + nt * 16 + nl;
    float bv = (bias && n < N) ? bias[n] : 0.f;
#pragma unroll
    for (int r = 0; r < 8; ++r) {
      int m = m0 + wave * 16 + r + ml;
      if (m < M && n < N) {
        float v = acc[nt][r] + bv;
        if (act == 1) v = gelu_exact(v);
        C[(size_t)m * N + n] = v;
      }
    }
  }
}

// ---------------------------------------------------------------------------
// f16-input GEMM with async-LDS staging (dominant vocab projection).
// A[M,K] f16, W[K,N] f16 (pre-converted, L2-resident), C f32 + bias.
// Requires M%128==0, N%128==0, K%32==0. Staging uses
// GLOBAL_LOAD_ASYNC_TO_LDS_B128 (ASYNCcnt) instead of VALU converts.
// ---------------------------------------------------------------------------
__global__ __launch_bounds__(256) void gemm_h16(
    const f16* __restrict__ A, const f16* __restrict__ W,
    const float* __restrict__ bias, float* __restrict__ C,
    int M, int N, int K)
{
  __shared__ __align__(16) f16 sA[128 * 32];   // [m][k], ld=32
  __shared__ __align__(16) f16 sB[32 * 128];   // [k][n], ld=128
  const int tid = threadIdx.x, wave = tid >> 5, lane = tid & 31;
  const int m0 = blockIdx.y * 128, n0 = blockIdx.x * 128;
  const unsigned sA0 = lds_off(sA), sB0 = lds_off(sB);
  v8f acc[8] = {};
  const int nk = K >> 5;
  for (int kt = 0; kt < nk; ++kt) {
    __syncthreads();
    // A slice: 128 rows x 32 halves = 512 x 16B chunks (chunk c: row c>>2, part c&3)
    // B slice: 32 rows x 128 halves = 512 x 16B chunks (chunk c: row c>>4, part c&15)
#pragma unroll
    for (int h = 0; h < 2; ++h) {
      int c = tid + h * 256;
      int ar = c >> 2, ap = c & 3;
      async_cp16(sA0 + (unsigned)((ar * 32 + ap * 8) * 2),
                 A + (size_t)(m0 + ar) * K + kt * 32 + ap * 8);
      int br = c >> 4, bp = c & 15;
      async_cp16(sB0 + (unsigned)((br * 128 + bp * 8) * 2),
                 W + (size_t)(kt * 32 + br) * N + n0 + bp * 8);
    }
    async_wait0();
    __syncthreads();
    v16h a = ld_frag_a(sA + (wave * 16 + (lane & 15)) * 32, lane >> 4);
#pragma unroll
    for (int nt = 0; nt < 8; ++nt) {
      v16h b = ld_frag_b(sB + (lane & 15) + nt * 16, 128, lane >> 4);
      acc[nt] = __builtin_amdgcn_wmma_f32_16x16x32_f16(
          false, a, false, b, (short)0, acc[nt], false, false);
    }
  }
  const int ml = 8 * (lane >> 4), nl = lane & 15;
#pragma unroll
  for (int nt = 0; nt < 8; ++nt) {
    int n = n0 + nt * 16 + nl;
    float bv = bias ? bias[n] : 0.f;
#pragma unroll
    for (int r = 0; r < 8; ++r) {
      int m = m0 + wave * 16 + r + ml;
      C[(size_t)m * N + n] = acc[nt][r] + bv;
    }
  }
}

// f32 -> f16 conversion (one-time weight downconvert; L2-resident afterwards)
__global__ __launch_bounds__(256) void k_cvt(const float* __restrict__ in,
                                             f16* __restrict__ out, int n)
{
  int i = blockIdx.x * 256 + threadIdx.x;
  if (i < n) out[i] = (f16)in[i];
}

// ---------------------------------------------------------------------------
// Task router: g_task = softmax(relu(h@W1+b1)@W2+b2); e_task from top-2 blend
// ---------------------------------------------------------------------------
__global__ __launch_bounds__(128) void k_task(
    const float* h_t, const float* W1, const float* b1,
    const float* W2, const float* b2, const float* cb,
    float* g_task, float* e_task)
{
  __shared__ float sh[256];
  __shared__ float hid[128];
  __shared__ float lg[3];
  __shared__ float sel[2];
  __shared__ int   sidx[2];
  const int b = blockIdx.x, t = threadIdx.x;
  sh[t] = h_t[b * 256 + t];
  sh[t + 128] = h_t[b * 256 + t + 128];
  __syncthreads();
  float a = b1[t];
  for (int d = 0; d < 256; ++d) a += sh[d] * W1[d * 128 + t];
  hid[t] = fmaxf(a, 0.f);
  __syncthreads();
  if (t < 3) {
    float s = b2[t];
    for (int h = 0; h < 128; ++h) s += hid[h] * W2[h * 3 + t];
    lg[t] = s;
  }
  __syncthreads();
  if (t == 0) {
    float m = fmaxf(lg[0], fmaxf(lg[1], lg[2]));
    float p[3];
    float s = 0.f;
    for (int i = 0; i < 3; ++i) { p[i] = expf(lg[i] - m); s += p[i]; }
    for (int i = 0; i < 3; ++i) { p[i] /= s; g_task[b * 3 + i] = p[i]; }
    int i0 = 0;
    if (p[1] > p[i0]) i0 = 1;
    if (p[2] > p[i0]) i0 = 2;
    int i1 = -1; float v1 = -1.f;
    for (int i = 0; i < 3; ++i) if (i != i0 && p[i] > v1) { v1 = p[i]; i1 = i; }
    sel[0] = p[i0]; sel[1] = v1; sidx[0] = i0; sidx[1] = i1;
  }
  __syncthreads();
  float tv0 = sel[0], tv1 = sel[1];
  int i0 = sidx[0], i1 = sidx[1];
  for (int j = t; j < 256; j += 128) {
    float c0 = cb[i0 * 256 + j];
    e_task[b * 256 + j] = (tv0 < 0.7f) ? (tv0 * c0 + tv1 * cb[i1 * 256 + j]) : c0;
  }
}

// ---------------------------------------------------------------------------
// Layout BERT embedding: x = LN(concat(prefix,h_t) + pos + type0)
// ---------------------------------------------------------------------------
__global__ __launch_bounds__(256) void k_embed_layout(
    const float* h_t, const float* prefix, const float* pos, const float* type0,
    const float* g, const float* bb, float* xout)
{
  const int r = blockIdx.x, b = r >> 1, s = r & 1, j = threadIdx.x;
  float x = (s == 0 ? prefix[j] : h_t[b * 256 + j]) + pos[s * 256 + j] + type0[j];
  __shared__ float red[8], red2[8];
  __shared__ float sm, si;
  const int lane = j & 31, w = j >> 5;
  float s1 = x, s2 = x * x;
  for (int off = 16; off; off >>= 1) { s1 += __shfl_xor(s1, off); s2 += __shfl_xor(s2, off); }
  if (lane == 0) { red[w] = s1; red2[w] = s2; }
  __syncthreads();
  if (j == 0) {
    float a = 0.f, c = 0.f;
    for (int i = 0; i < 8; ++i) { a += red[i]; c += red2[i]; }
    float mean = a * (1.f / 256.f);
    float var = c * (1.f / 256.f) - mean * mean;
    sm = mean; si = rsqrtf(var + 1e-12f);
  }
  __syncthreads();
  xout[(size_t)r * 256 + j] = (x - sm) * si * g[j] + bb[j];
}

// residual + layernorm: out = LN(xin + add) * g + bb (in-place safe)
__global__ __launch_bounds__(256) void k_resln(
    const float* xin, const float* add, const float* g, const float* bb, float* xout)
{
  const int r = blockIdx.x, j = threadIdx.x;
  float x = xin[(size_t)r * 256 + j] + add[(size_t)r * 256 + j];
  __shared__ float red[8], red2[8];
  __shared__ float sm, si;
  const int lane = j & 31, w = j >> 5;
  float s1 = x, s2 = x * x;
  for (int off = 16; off; off >>= 1) { s1 += __shfl_xor(s1, off); s2 += __shfl_xor(s2, off); }
  if (lane == 0) { red[w] = s1; red2[w] = s2; }
  __syncthreads();
  if (j == 0) {
    float a = 0.f, c = 0.f;
    for (int i = 0; i < 8; ++i) { a += red[i]; c += red2[i]; }
    float mean = a * (1.f / 256.f);
    float var = c * (1.f / 256.f) - mean * mean;
    sm = mean; si = rsqrtf(var + 1e-12f);
  }
  __syncthreads();
  xout[(size_t)r * 256 + j] = (x - sm) * si * g[j] + bb[j];
}

// Layout attention, S=2, NH=8, dh=32, one block per sample
__global__ __launch_bounds__(64) void k_attn2(
    const float* q, const float* k, const float* v, float* ctx)
{
  const int b = blockIdx.x, t = threadIdx.x;
  __shared__ float sc[32];  // [h][i][j]
  if (t < 32) {
    int h = t >> 2, i = (t >> 1) & 1, j = t & 1;
    const float* qp = q + (size_t)(b * 2 + i) * 256 + h * 32;
    const float* kp = k + (size_t)(b * 2 + j) * 256 + h * 32;
    float s = 0.f;
    for (int d = 0; d < 32; ++d) s += qp[d] * kp[d];
    sc[t] = s * 0.1767766952966369f;  // 1/sqrt(32)
  }
  __syncthreads();
  if (t < 16) {
    int h = t >> 1, i = t & 1;
    float a0 = sc[h * 4 + i * 2 + 0], a1 = sc[h * 4 + i * 2 + 1];
    float m = fmaxf(a0, a1);
    float e0 = expf(a0 - m), e1 = expf(a1 - m);
    float s = e0 + e1;
    sc[h * 4 + i * 2 + 0] = e0 / s;
    sc[h * 4 + i * 2 + 1] = e1 / s;
  }
  __syncthreads();
  for (int e = t; e < 512; e += 64) {
    int i = e >> 8, c = e & 255, h = c >> 5;
    float a0 = sc[h * 4 + i * 2 + 0], a1 = sc[h * 4 + i * 2 + 1];
    ctx[(size_t)(b * 2 + i) * 256 + c] =
        a0 * v[(size_t)(b * 2 + 0) * 256 + c] + a1 * v[(size_t)(b * 2 + 1) * 256 + c];
  }
}

// Layout head: llog, lprobs, pred, e_layout
__global__ __launch_bounds__(64) void k_layout_head(
    const float* x, const int* prev_layout, const float* Wp, const float* bp,
    const float* cb, float* lprobs_out, float* pred_out, float* e_layout, int* pred_ws)
{
  const int b = blockIdx.x, t = threadIdx.x;
  __shared__ float lg[5];
  __shared__ int spred;
  if (t < 5) {
    float s = bp[t];
    const float* xr = x + (size_t)(b * 2) * 256;
    for (int d = 0; d < 256; ++d) s += xr[d] * Wp[d * 5 + t];
    if (prev_layout[b] == t) s += 0.3f;
    lg[t] = s;
  }
  __syncthreads();
  if (t == 0) {
    float m = lg[0];
    for (int i = 1; i < 5; ++i) m = fmaxf(m, lg[i]);
    float e[5], s = 0.f;
    for (int i = 0; i < 5; ++i) { e[i] = expf(lg[i] - m); s += e[i]; }
    int am = 0; float mx = -1.f;
    for (int i = 0; i < 5; ++i) {
      float p = e[i] / s;
      lprobs_out[b * 5 + i] = p;
      if (p > mx) { mx = p; am = i; }
    }
    int pred = (mx < 0.8f) ? prev_layout[b] : am;
    pred_out[b] = (float)pred;
    pred_ws[b] = pred;
    spred = pred;
  }
  __syncthreads();
  for (int j = t; j < 256; j += 64) e_layout[b * 256 + j] = cb[spred * 256 + j];
}

__global__ void k_init(int* flag) {
  if (threadIdx.x == 0 && blockIdx.x == 0) *flag = 1;
}

// MoE gating
__global__ __launch_bounds__(128) void k_gate(
    const float* h_t, const float* e_task, const float* e_layout,
    const float* Wr1, const float* br1, const float* Wr2, const float* br2,
    const float* Wh, const float* bh,
    float* gp_out, float* gv_ws, int* gi_ws, int* flag)
{
  __shared__ float fl[768];
  __shared__ float hid[128];
  __shared__ float lg[5];
  const int b = blockIdx.x, t = threadIdx.x;
  for (int j = t; j < 256; j += 128) {
    fl[j] = h_t[b * 256 + j];
    fl[256 + j] = e_task[b * 256 + j];
    fl[512 + j] = e_layout[b * 256 + j];
  }
  __syncthreads();
  float a = br1[t];
  for (int d = 0; d < 768; ++d) a += fl[d] * Wr1[d * 128 + t];
  hid[t] = fmaxf(a, 0.f);
  __syncthreads();
  if (t < 5) {
    float s = br2[t] + bh[t];
    for (int h = 0; h < 128; ++h) s += hid[h] * Wr2[h * 5 + t];
    for (int d = 0; d < 256; ++d) s += fl[512 + d] * Wh[d * 5 + t];
    lg[t] = s;
  }
  __syncthreads();
  if (t == 0) {
    float m = lg[0];
    for (int i = 1; i < 5; ++i) m = fmaxf(m, lg[i]);
    float e[5], s = 0.f;
    for (int i = 0; i < 5; ++i) { e[i] = expf(lg[i] - m); s += e[i]; }
    float p[5];
    for (int i = 0; i < 5; ++i) { p[i] = e[i] / s; gp_out[b * 5 + i] = p[i]; }
    int i0 = 0;
    for (int i = 1; i < 5; ++i) if (p[i] > p[i0]) i0 = i;
    int i1 = -1; float v1 = -1.f;
    for (int i = 0; i < 5; ++i) if (i != i0 && p[i] > v1) { v1 = p[i]; i1 = i; }
    gv_ws[b * 2 + 0] = p[i0];
    gv_ws[b * 2 + 1] = v1;
    gi_ws[b * 2 + 0] = i0;
    gi_ws[b * 2 + 1] = i1;
    if (!(p[i0] > 0.7f)) atomicAnd(flag, 0);
  }
}

// Build x_t tokens [B,8,256]
__global__ __launch_bounds__(256) void k_build_xt(
    const float* h_t, const float* e_task, const float* e_layout,
    const int* prev_tokens, const float* token_emb, float* xt)
{
  const int r = blockIdx.x, b = r >> 3, s = r & 7, j = threadIdx.x;
  float v;
  if (s == 0) v = h_t[b * 256 + j];
  else if (s == 1) v = e_task[b * 256 + j];
  else if (s == 2) v = e_layout[b * 256 + j];
  else {
    int tok = prev_tokens[b * 5 + (s - 3)];
    v = token_emb[(size_t)tok * 256 + j];
  }
  xt[(size_t)r * 256 + j] = v;
}

// ---------------------------------------------------------------------------
// MoE fused expert block helpers
// ---------------------------------------------------------------------------
__device__ __forceinline__ void moe_gemm(
    const f16* sAm, int Ka, const float* Wg, int ldw, int ncol0,
    const float* biasg, int act, float* outf, f16* stage)
{
  const int tid = threadIdx.x, wave = tid >> 5, lane = tid & 31;
  v8f acc0 = {}, acc1 = {};
  const int nk = Ka >> 5;
  for (int kt = 0; kt < nk; ++kt) {
    __syncthreads();
    for (int e = tid; e < 32 * 256; e += 256) {
      int r = e >> 8, c = e & 255;
      stage[e] = (f16)Wg[(size_t)(kt * 32 + r) * ldw + ncol0 + c];
    }
    __syncthreads();
    v16h a = ld_frag_a(sAm + (lane & 15) * Ka + kt * 32, lane >> 4);
    v16h b0 = ld_frag_b(stage + (lane & 15) + (wave * 2 + 0) * 16, 256, lane >> 4);
    v16h b1 = ld_frag_b(stage + (lane & 15) + (wave * 2 + 1) * 16, 256, lane >> 4);
    acc0 = __builtin_amdgcn_wmma_f32_16x16x32_f16(false, a, false, b0, (short)0, acc0, false, false);
    acc1 = __builtin_amdgcn_wmma_f32_16x16x32_f16(false, a, false, b1, (short)0, acc1, false, false);
  }
  __syncthreads();
  if (lane < 16) {  // these lanes hold m = 0..7 (valid rows)
    int n0 = (wave * 2 + 0) * 16 + lane;
    int n1 = (wave * 2 + 1) * 16 + lane;
    float bv0 = biasg ? biasg[ncol0 + n0] : 0.f;
    float bv1 = biasg ? biasg[ncol0 + n1] : 0.f;
#pragma unroll
    for (int r = 0; r < 8; ++r) {
      float v0 = acc0[r] + bv0;
      float v1 = acc1[r] + bv1;
      if (act == 1) { v0 = gelu_exact(v0); v1 = gelu_exact(v1); }
      outf[r * 256 + n0] = v0;
      outf[r * 256 + n1] = v1;
    }
  }
  __syncthreads();
}

// layernorm over 8 rows of 256: wave w handles row w.
__device__ __forceinline__ void ln8(
    const float* xin, const float* xadd, const float* g, const float* bb,
    float* outf, f16* outh, float* zacc, float wgt)
{
  const int wave = threadIdx.x >> 5, lane = threadIdx.x & 31;
  float vals[8];
  float s1 = 0.f, s2 = 0.f;
#pragma unroll
  for (int i = 0; i < 8; ++i) {
    int c = lane + 32 * i;
    float x = xin[wave * 256 + c] + (xadd ? xadd[wave * 256 + c] : 0.f);
    vals[i] = x; s1 += x; s2 += x * x;
  }
#pragma unroll
  for (int off = 16; off; off >>= 1) { s1 += __shfl_xor(s1, off); s2 += __shfl_xor(s2, off); }
  float mean = s1 * (1.f / 256.f);
  float var = s2 * (1.f / 256.f) - mean * mean;
  float inv = rsqrtf(var + 1e-12f);
#pragma unroll
  for (int i = 0; i < 8; ++i) {
    int c = lane + 32 * i;
    float y = (vals[i] - mean) * inv * g[c] + bb[c];
    if (outf) outf[wave * 256 + c] = y;
    if (outh) outh[wave * 256 + c] = (f16)y;
    if (zacc) zacc[wave * 256 + c] += wgt * y;
  }
  __syncthreads();
}

// Fused MoE: per sample, run the 2 selected experts (embed + 1 BERT layer),
// accumulate z = w0*expert(gi0) + w1*expert(gi1), emit z in f16 for the
// vocab GEMM. 256 threads = 8 waves.
__global__ __launch_bounds__(256) void k_moe(
    const float* xt, const float* gv, const int* gi, const int* flag,
    const float* pos_e, const float* type0_e, const float* lneg_e, const float* lneb_e,
    const float* Wq_e, const float* bq_e, const float* Wk_e, const float* bk_e,
    const float* Wv_e, const float* bv_e, const float* Wo_e, const float* bo_e,
    const float* ln1g_e, const float* ln1b_e,
    const float* W1_e, const float* b1_e, const float* W2_e, const float* b2_e,
    const float* ln2g_e, const float* ln2b_e,
    f16* z)
{
  __shared__ __align__(16) f16 sA[16 * 256];     // activation tile (rows 8..15 zero)
  __shared__ __align__(16) f16 sH[16 * 1024];    // FFN hidden tile (rows 8..15 zero)
  __shared__ __align__(16) f16 sStage[32 * 256]; // weight staging
  __shared__ float sXemb[8 * 256];
  __shared__ float sX1[8 * 256];
  __shared__ float sQ[8 * 256];
  __shared__ float sKv[8 * 256];
  __shared__ float sVv[8 * 256];
  __shared__ float sTmp[8 * 256];
  __shared__ float sZ[8 * 256];
  __shared__ float sSc[8 * 8 * 8];
  const int b = blockIdx.x, tid = threadIdx.x;

  for (int i = tid; i < 8 * 256; i += 256) sA[8 * 256 + i] = (f16)0.f;
  for (int i = tid; i < 8 * 1024; i += 256) sH[8 * 1024 + i] = (f16)0.f;
  for (int i = tid; i < 8 * 256; i += 256) sZ[i] = 0.f;

  const float w0 = gv[b * 2 + 0];
  const float w1 = (*flag) ? 0.f : gv[b * 2 + 1];

  for (int slot = 0; slot < 2; ++slot) {
    const int ex = gi[b * 2 + slot];
    const float wgt = (slot == 0) ? w0 : w1;
    const size_t eD = (size_t)ex * 256;
    const size_t eDD = (size_t)ex * 256 * 256;
    const size_t eDF = (size_t)ex * 256 * 1024;

    // embed raw: xt + pos + type0
    for (int i = tid; i < 8 * 256; i += 256) {
      int s = i >> 8, c = i & 255;
      sXemb[i] = xt[(size_t)b * 2048 + i] + pos_e[(size_t)ex * 512 * 256 + s * 256 + c]
               + type0_e[eD + c];
    }
    __syncthreads();
    ln8(sXemb, nullptr, lneg_e + eD, lneb_e + eD, sXemb, sA, nullptr, 0.f);

    // Q, K, V
    moe_gemm(sA, 256, Wq_e + eDD, 256, 0, bq_e + eD, 0, sQ, sStage);
    moe_gemm(sA, 256, Wk_e + eDD, 256, 0, bk_e + eD, 0, sKv, sStage);
    moe_gemm(sA, 256, Wv_e + eDD, 256, 0, bv_e + eD, 0, sVv, sStage);

    // attention S=8, NH=8, dh=32
    for (int i2 = tid; i2 < 512; i2 += 256) {
      int h = i2 >> 6, i = (i2 >> 3) & 7, jj = i2 & 7;
      const float* qp = sQ + i * 256 + h * 32;
      const float* kp = sKv + jj * 256 + h * 32;
      float s = 0.f;
      for (int d = 0; d < 32; ++d) s += qp[d] * kp[d];
      sSc[i2] = s * 0.1767766952966369f;
    }
    __syncthreads();
    if (tid < 64) {
      float* row = sSc + tid * 8;
      float m = row[0];
      for (int i = 1; i < 8; ++i) m = fmaxf(m, row[i]);
      float ssum = 0.f;
      for (int i = 0; i < 8; ++i) { row[i] = expf(row[i] - m); ssum += row[i]; }
      float inv = 1.f / ssum;
      for (int i = 0; i < 8; ++i) row[i] *= inv;
    }
    __syncthreads();
    for (int i2 = tid; i2 < 2048; i2 += 256) {
      int s = i2 >> 8, c = i2 & 255, h = c >> 5;
      const float* arow = sSc + (h * 8 + s) * 8;
      float acc = 0.f;
      for (int jj = 0; jj < 8; ++jj) acc += arow[jj] * sVv[jj * 256 + c];
      sTmp[i2] = acc;
    }
    __syncthreads();
    for (int i2 = tid; i2 < 2048; i2 += 256) sA[i2] = (f16)sTmp[i2];
    __syncthreads();

    // o = ctx@Wo + bo ; x1 = LN1(xemb + o)
    moe_gemm(sA, 256, Wo_e + eDD, 256, 0, bo_e + eD, 0, sQ, sStage);
    ln8(sXemb, sQ, ln1g_e + eD, ln1b_e + eD, sX1, sA, nullptr, 0.f);

    // FFN: h = gelu(x1@W1+b1)  (N=1024 in 4 chunks of 256)
    for (int nc = 0; nc < 4; ++nc) {
      moe_gemm(sA, 256, W1_e + eDF, 1024, nc * 256, b1_e + (size_t)ex * 1024, 1, sTmp, sStage);
      for (int i2 = tid; i2 < 2048; i2 += 256) {
        int s = i2 >> 8, c = i2 & 255;
        sH[s * 1024 + nc * 256 + c] = (f16)sTmp[i2];
      }
      __syncthreads();
    }
    // out = h@W2 + b2 ; z += wgt * LN2(x1 + out)
    moe_gemm(sH, 1024, W2_e + eDF, 256, 0, b2_e + eD, 0, sTmp, sStage);
    ln8(sX1, sTmp, ln2g_e + eD, ln2b_e + eD, nullptr, nullptr, sZ, wgt);
  }

  for (int i2 = tid; i2 < 2048; i2 += 256)
    z[(size_t)b * 2048 + i2] = (f16)sZ[i2];
}

// ---------------------------------------------------------------------------
// Host-side launcher
// ---------------------------------------------------------------------------
extern "C" void kernel_launch(void* const* d_in, const int* in_sizes, int n_in,
                              void* d_out, int out_size, void* d_ws, size_t ws_size,
                              hipStream_t stream)
{
  (void)in_sizes; (void)n_in; (void)out_size; (void)ws_size;
#define FIN(i) ((const float*)d_in[(i)])
#define IIN(i) ((const int*)d_in[(i)])
  const float* h_t = FIN(0);
  const int* prev_layout = IIN(1);
  const int* prev_tokens = IIN(2);
  // task_params
  const float *tW1 = FIN(3), *tb1 = FIN(4), *tW2 = FIN(5), *tb2 = FIN(6), *tcb = FIN(7);
  // layout_params
  const float *prefix = FIN(8), *lpos = FIN(9), *ltype0 = FIN(10), *lneg = FIN(11), *lneb = FIN(12);
  // layers at 13 + l*16 (Wq,bq,Wk,bk,Wv,bv,Wo,bo,ln1_g,ln1_b,W1,b1,W2,b2,ln2_g,ln2_b)
  const float *Wp = FIN(45), *bp = FIN(46), *lcb = FIN(47);
  // moe_params
  const float *Wr1 = FIN(48), *br1 = FIN(49), *Wr2 = FIN(50), *br2 = FIN(51);
  const float *Wh = FIN(52), *bh = FIN(53);
  const float *epos = FIN(54), *etype0 = FIN(55), *elneg = FIN(56), *elneb = FIN(57);
  const float *eWq = FIN(58), *ebq = FIN(59), *eWk = FIN(60), *ebk = FIN(61);
  const float *eWv = FIN(62), *ebv = FIN(63), *eWo = FIN(64), *ebo = FIN(65);
  const float *eln1g = FIN(66), *eln1b = FIN(67);
  const float *eW1 = FIN(68), *eb1 = FIN(69), *eW2 = FIN(70), *eb2 = FIN(71);
  const float *eln2g = FIN(72), *eln2b = FIN(73);
  const float *Wout = FIN(74), *bout = FIN(75), *token_emb = FIN(76);

  float* out = (float*)d_out;
  float* o_logits = out;                                      // [512*8, 32000]
  float* o_gtask  = out + (size_t)4096 * 32000;               // [512,3]
  float* o_pred   = o_gtask + 512 * 3;                        // [512]
  float* o_lprobs = o_pred + 512;                             // [512,5]
  float* o_gp     = o_lprobs + 512 * 5;                       // [512,5]

  float* ws = (float*)d_ws;
  float* w_etask = ws;      ws += 512 * 256;
  float* w_elayout = ws;    ws += 512 * 256;
  float* w_x = ws;          ws += 1024 * 256;
  float* w_q = ws;          ws += 1024 * 256;
  float* w_k = ws;          ws += 1024 * 256;
  float* w_v = ws;          ws += 1024 * 256;
  float* w_ctx = ws;        ws += 1024 * 256;
  float* w_h = ws;          ws += 1024 * 1024;
  float* w_xt = ws;         ws += 4096 * 256;
  float* w_gv = ws;         ws += 1024;
  int* w_gi = (int*)ws;     ws += 1024;
  int* w_pred = (int*)ws;   ws += 512;
  int* w_flag = (int*)ws;   ws += 16;
  f16* w_z16 = (f16*)ws;    ws += (4096 * 256) / 2;           // z in f16
  f16* w_wout16 = (f16*)ws; ws += (256 * 32000) / 2;          // Wout in f16

  // ---- One-time weight downconvert for the vocab GEMM (L2-resident) ----
  k_cvt<<<(256 * 32000 + 255) / 256, 256, 0, stream>>>(Wout, w_wout16, 256 * 32000);

  // ---- Task router ----
  k_task<<<512, 128, 0, stream>>>(h_t, tW1, tb1, tW2, tb2, tcb, o_gtask, w_etask);

  // ---- Layout router: 2-layer BERT on [prefix, h_t] ----
  k_embed_layout<<<1024, 256, 0, stream>>>(h_t, prefix, lpos, ltype0, lneg, lneb, w_x);
  for (int l = 0; l < 2; ++l) {
    int base = 13 + l * 16;
    const float *Wq = FIN(base + 0), *bq = FIN(base + 1);
    const float *Wk = FIN(base + 2), *bk = FIN(base + 3);
    const float *Wv = FIN(base + 4), *bv = FIN(base + 5);
    const float *Wo = FIN(base + 6), *bo = FIN(base + 7);
    const float *g1 = FIN(base + 8), *be1 = FIN(base + 9);
    const float *W1 = FIN(base + 10), *bb1 = FIN(base + 11);
    const float *W2 = FIN(base + 12), *bb2 = FIN(base + 13);
    const float *g2 = FIN(base + 14), *be2 = FIN(base + 15);
    dim3 g256(2, 8);   // N=256, M=1024
    dim3 gffn(8, 8);   // N=1024, M=1024
    gemm128<<<g256, 256, 0, stream>>>(w_x, Wq, bq, w_q, 1024, 256, 256, 0);
    gemm128<<<g256, 256, 0, stream>>>(w_x, Wk, bk, w_k, 1024, 256, 256, 0);
    gemm128<<<g256, 256, 0, stream>>>(w_x, Wv, bv, w_v, 1024, 256, 256, 0);
    k_attn2<<<512, 64, 0, stream>>>(w_q, w_k, w_v, w_ctx);
    gemm128<<<g256, 256, 0, stream>>>(w_ctx, Wo, bo, w_q, 1024, 256, 256, 0);
    k_resln<<<1024, 256, 0, stream>>>(w_x, w_q, g1, be1, w_x);
    gemm128<<<gffn, 256, 0, stream>>>(w_x, W1, bb1, w_h, 1024, 1024, 256, 1);
    gemm128<<<g256, 256, 0, stream>>>(w_h, W2, bb2, w_q, 1024, 256, 1024, 0);
    k_resln<<<1024, 256, 0, stream>>>(w_x, w_q, g2, be2, w_x);
  }
  k_layout_head<<<512, 64, 0, stream>>>(w_x, prev_layout, Wp, bp, lcb,
                                        o_lprobs, o_pred, w_elayout, w_pred);

  // ---- MoE gating ----
  k_init<<<1, 32, 0, stream>>>(w_flag);
  k_gate<<<512, 128, 0, stream>>>(h_t, w_etask, w_elayout, Wr1, br1, Wr2, br2,
                                  Wh, bh, o_gp, w_gv, w_gi, w_flag);

  // ---- MoE experts (z emitted in f16) ----
  k_build_xt<<<4096, 256, 0, stream>>>(h_t, w_etask, w_elayout, prev_tokens, token_emb, w_xt);
  k_moe<<<512, 256, 0, stream>>>(w_xt, w_gv, w_gi, w_flag, epos, etype0, elneg, elneb,
                                 eWq, ebq, eWk, ebk, eWv, ebv, eWo, ebo, eln1g, eln1b,
                                 eW1, eb1, eW2, eb2, eln2g, eln2b, w_z16);

  // ---- Vocabulary projection (dominant GEMM, async-LDS staged f16) ----
  dim3 glog(250, 32);  // N=32000, M=4096
  gemm_h16<<<glog, 256, 0, stream>>>(w_z16, w_wout16, bout, o_logits, 4096, 32000, 256);
#undef FIN
#undef IIN
}
